// GatedGCNLayer_6700148982559
// MI455X (gfx1250) — matrix operand order repaired and checked by reading, not compiled, not run on hardware
//
#include <hip/hip_runtime.h>
#include <hip/hip_bf16.h>
#include <math.h>

#define NN    10000
#define NPAD  10112            // 79 * 128, padded row count for node ws buffers
#define DEGC  64
#define EE    (NN * DEGC)
#define DDIM  128

typedef __bf16 v16bf __attribute__((ext_vector_type(16)));
typedef __bf16 v8bf  __attribute__((ext_vector_type(8)));
typedef float  v8f   __attribute__((ext_vector_type(8)));

__device__ __forceinline__ unsigned short f2bf(float f) {
    unsigned int u = __float_as_uint(f);
    u += 0x7FFFu + ((u >> 16) & 1u);          // round-to-nearest-even
    return (unsigned short)(u >> 16);
}

__device__ __forceinline__ v16bf frag16(const unsigned short* p0, const unsigned short* p1) {
    v8bf lo = *reinterpret_cast<const v8bf*>(p0);
    v8bf hi = *reinterpret_cast<const v8bf*>(p1);
    return __builtin_shufflevector(lo, hi, 0,1,2,3,4,5,6,7,8,9,10,11,12,13,14,15);
}

__device__ __forceinline__ float getc(const float4& v, int j) {
    return j == 0 ? v.x : (j == 1 ? v.y : (j == 2 ? v.z : v.w));
}

// gfx1250 async copy: global -> LDS, one 16B transaction, tracked by ASYNCcnt
__device__ __forceinline__ void async_load_b128(unsigned ldsOff, const void* gptr) {
    asm volatile("global_load_async_to_lds_b128 %0, %1, off"
                 :: "v"(ldsOff), "v"((unsigned long long)(size_t)gptr)
                 : "memory");
}
__device__ __forceinline__ void wait_async0() {
    asm volatile("s_wait_asynccnt 0" ::: "memory");
}
__device__ __forceinline__ void wait_async8() {   // oldest 8 retired (in-order)
    asm volatile("s_wait_asynccnt 8" ::: "memory");
}

struct GemmJob  { const unsigned short* B; const float* bias; float* C; };
struct GemmJobs { GemmJob j[5]; };
struct Ptr8     { const float* p[8]; };

// ---------------------------------------------------------------------------
// One-shot weight prep: f32 [128x128] row-major -> bf16, transposed [n][k],
// XOR-swizzled in 16-elem k-blocks (phys_blk = (k>>4) ^ (n&7)).
// After this, every GEMM B tile is a LINEAR 32KB async copy into LDS.
// ---------------------------------------------------------------------------
__global__ __launch_bounds__(256) void pack_weights(Ptr8 wsrc, unsigned short* out)
{
    const float* W = wsrc.p[blockIdx.x];
    unsigned short* o = out + (size_t)blockIdx.x * 16384;
#pragma unroll 8
    for (int i = 0; i < 64; ++i) {
        int idx = threadIdx.x + i * 256;
        int k = idx >> 7, n = idx & 127;
        int blk = (k >> 4) ^ (n & 7);
        o[n * 128 + blk * 16 + (k & 15)] = f2bf(W[k * 128 + n]);
    }
}

// ---------------------------------------------------------------------------
// Multi-job GEMM: for each job, C[M x 128] = A[M x 128] * B + bias.
// Block = 128 rows x 128 cols; A read from HBM once, converted to bf16 LDS.
// B tiles are pre-swizzled bf16 -> double-buffered async-to-LDS copies that
// overlap WMMA compute (s_wait_asynccnt 8 pipeline). Stores are unguarded
// (outputs padded to a multiple of 128 rows); A row loads clamp instead.
// ---------------------------------------------------------------------------
template<int NJOBS>
__global__ __launch_bounds__(256) void gemm128_multi(
    const float* __restrict__ A, GemmJobs jobs, int M)
{
    __shared__ __align__(16) unsigned short As[128 * 128];      // 32 KB
    __shared__ __align__(16) unsigned short Bt[2][128 * 128];   // 64 KB

    const int tid     = threadIdx.x;
    const int rowBase = blockIdx.x * 128;

    // prologue: async copy of job 0 B tile (linear 32KB)
    const unsigned btb[2] = { (unsigned)(size_t)(const void*)&Bt[0][0],
                              (unsigned)(size_t)(const void*)&Bt[1][0] };
    {
        const char* Bw = (const char*)jobs.j[0].B;
#pragma unroll
        for (int i = 0; i < 8; ++i) {
            int idx = tid + i * 256;
            async_load_b128(btb[0] + (unsigned)idx * 16, Bw + (size_t)idx * 16);
        }
    }

    // A tile (f32 -> bf16), swizzled; clamp row (padded outputs make it safe)
#pragma unroll
    for (int i = 0; i < 16; ++i) {
        int idx4 = tid + i * 256;           // 0..4095 float4 granules
        int r    = idx4 >> 5;               // 0..127
        int c4   = idx4 & 31;               // k = c4*4 .. c4*4+3
        int rg   = rowBase + r;
        if (rg >= M) rg = M - 1;
        float4 v = reinterpret_cast<const float4*>(A)[rg * 32 + c4];
        int blk = (c4 >> 2) ^ (r & 7);
        uint2 pk;
        pk.x = (unsigned int)f2bf(v.x) | ((unsigned int)f2bf(v.y) << 16);
        pk.y = (unsigned int)f2bf(v.z) | ((unsigned int)f2bf(v.w) << 16);
        *reinterpret_cast<uint2*>(&As[r * 128 + blk * 16 + (c4 & 3) * 4]) = pk;
    }

    const int w     = tid >> 5;
    const int l     = tid & 31;
    const int aRow  = (w << 4) + (l & 15);
    const int khalf = l >> 4;
    const int rbase = rowBase + (w << 4) + (khalf << 3);

    v16bf aF[4];
#pragma unroll
    for (int jj = 0; jj < NJOBS; ++jj) {
        if (jj + 1 < NJOBS) {               // prefetch next B while computing
            const char* Bw = (const char*)jobs.j[jj + 1].B;
            unsigned dstb = btb[(jj + 1) & 1];
#pragma unroll
            for (int i = 0; i < 8; ++i) {
                int idx = tid + i * 256;
                async_load_b128(dstb + (unsigned)idx * 16, Bw + (size_t)idx * 16);
            }
            wait_async8();                  // current job's 8 txns retired
        } else {
            wait_async0();
        }
        __syncthreads();

        if (jj == 0) {                      // A fragments: load once, reuse
#pragma unroll
            for (int kk = 0; kk < 4; ++kk) {
                int b0 = (kk * 2)     ^ (aRow & 7);
                int b1 = (kk * 2 + 1) ^ (aRow & 7);
                aF[kk] = frag16(&As[aRow * 128 + b0 * 16 + khalf * 8],
                                &As[aRow * 128 + b1 * 16 + khalf * 8]);
            }
        }

        const unsigned short* Bcur = &Bt[jj & 1][0];
        const float* bias = jobs.j[jj].bias;    // never null (zero vector ws)
        float*       C    = jobs.j[jj].C;
#pragma unroll
        for (int t = 0; t < 8; ++t) {
            const int col = (t << 4) + (l & 15);
            v8f acc = {0.f, 0.f, 0.f, 0.f, 0.f, 0.f, 0.f, 0.f};
#pragma unroll
            for (int kk = 0; kk < 4; ++kk) {
                int pb0 = (kk * 2)     ^ (col & 7);
                int pb1 = (kk * 2 + 1) ^ (col & 7);
                v16bf b = frag16(&Bcur[col * 128 + pb0 * 16 + khalf * 8],
                                 &Bcur[col * 128 + pb1 * 16 + khalf * 8]);
                acc = __builtin_amdgcn_wmma_f32_16x16x32_bf16(
                          false, aF[kk], false, b, (short)0, acc, false, false);
            }
            const float bb = bias[col];
#pragma unroll
            for (int v = 0; v < 8; ++v)
                C[(rbase + v) * 128 + col] = acc[v] + bb;
        }
        if (jj + 1 < NJOBS) __syncthreads();    // before overwriting Bt
    }
}

// ---------------------------------------------------------------------------
// Fused: sigma = sigmoid(e_new)@W_eta  +  segmented (num, den) reduction.
// A (bf16 activations) and B (pre-swizzled W_eta) both arrive via
// global_load_async_to_lds_b128. dst = repeat(arange(N),64): 128 rows/block
// == exactly 2 nodes -> in-LDS reduction (reusing A tile), no global atomics.
// ---------------------------------------------------------------------------
__global__ __launch_bounds__(256) void sigma_gemm_reduce(
    const unsigned short* __restrict__ sigA,  // [E x 128] bf16
    const unsigned short* __restrict__ Bsw,   // W_eta pre-swizzled bf16
    const float* __restrict__ Vh, const int* __restrict__ src,
    float* __restrict__ numO, float* __restrict__ denO)
{
    __shared__ __align__(16) unsigned short As[128 * 128];
    __shared__ __align__(16) unsigned short Bt[128 * 128];

    const int tid     = threadIdx.x;
    const int rowBase = blockIdx.x * 128;

    const unsigned abase = (unsigned)(size_t)(const void*)As;
    const unsigned bbase = (unsigned)(size_t)(const void*)Bt;
    const char*    ga    = (const char*)sigA + (size_t)rowBase * 256;  // 256 B/row
    const char*    gb    = (const char*)Bsw;
#pragma unroll
    for (int i = 0; i < 8; ++i) {
        int idx = tid + i * 256;            // 16B units
        int r = idx >> 4, c8 = idx & 15;
        unsigned phys = (unsigned)((c8 >> 1) ^ (r & 7));
        unsigned loff = abase + (unsigned)(r * 256 + phys * 32 + (c8 & 1) * 16);
        async_load_b128(loff, ga + (size_t)idx * 16);
    }
#pragma unroll
    for (int i = 0; i < 8; ++i) {
        int idx = tid + i * 256;
        async_load_b128(bbase + (unsigned)idx * 16, gb + (size_t)idx * 16);
    }
    wait_async0();
    __syncthreads();

    const int w     = tid >> 5;
    const int l     = tid & 31;
    const int aRow  = (w << 4) + (l & 15);
    const int khalf = l >> 4;
    const int rbase = rowBase + (w << 4) + (khalf << 3);

    v16bf aF[4];
#pragma unroll
    for (int kk = 0; kk < 4; ++kk) {
        int b0 = (kk * 2)     ^ (aRow & 7);
        int b1 = (kk * 2 + 1) ^ (aRow & 7);
        aF[kk] = frag16(&As[aRow * 128 + b0 * 16 + khalf * 8],
                        &As[aRow * 128 + b1 * 16 + khalf * 8]);
    }
    int sv[8];
#pragma unroll
    for (int v = 0; v < 8; ++v) sv[v] = src[rbase + v];

    float pn[8], pd[8];
#pragma unroll
    for (int t = 0; t < 8; ++t) {
        const int col = (t << 4) + (l & 15);
        v8f acc = {0.f, 0.f, 0.f, 0.f, 0.f, 0.f, 0.f, 0.f};
#pragma unroll
        for (int kk = 0; kk < 4; ++kk) {
            int pb0 = (kk * 2)     ^ (col & 7);
            int pb1 = (kk * 2 + 1) ^ (col & 7);
            v16bf b = frag16(&Bt[col * 128 + pb0 * 16 + khalf * 8],
                             &Bt[col * 128 + pb1 * 16 + khalf * 8]);
            acc = __builtin_amdgcn_wmma_f32_16x16x32_bf16(
                      false, aF[kk], false, b, (short)0, acc, false, false);
        }
        float a0 = 0.f, a1 = 0.f;
#pragma unroll
        for (int v = 0; v < 8; ++v) {
            float s = acc[v];
            a0 += s * Vh[sv[v] * 128 + col];
            a1 += s;
        }
        pn[t] = a0; pd[t] = a1;
    }

    // reuse As LDS as reduction scratch: [0:256) num, [256:512) den (floats)
    __syncthreads();
    float* red = reinterpret_cast<float*>(As);
    red[tid] = 0.f; red[tid + 256] = 0.f;
    __syncthreads();
    const int nodeLocal = w >> 2;           // waves 0-3: node0, 4-7: node1
#pragma unroll
    for (int t = 0; t < 8; ++t) {
        const int col = (t << 4) + (l & 15);
        atomicAdd(&red[nodeLocal * 128 + col], pn[t]);
        atomicAdd(&red[256 + nodeLocal * 128 + col], pd[t]);
    }
    __syncthreads();
    {
        int nl = tid >> 7, cc = tid & 127;
        int node = blockIdx.x * 2 + nl;
        numO[node * 128 + cc] = red[tid];
        denO[node * 128 + cc] = red[256 + tid];
    }
}

// ---------------------------------------------------------------------------
// Elementwise / statistics kernels
// ---------------------------------------------------------------------------
__global__ void zero_buf(float* p, int n) {
    int idx = blockIdx.x * blockDim.x + threadIdx.x;
    if (idx < n) p[idx] = 0.f;
}

// stats layout: [0:128) e-sum, [128:256) e-sumsq, [256:384) h-sum, [384:512) h-sumsq
__global__ __launch_bounds__(256) void edge_stats(
    const float* __restrict__ W1e, const float* __restrict__ W2h,
    const float* __restrict__ W3h, const int* __restrict__ src,
    const int* __restrict__ dst, float* __restrict__ stats)
{
    const int cg = threadIdx.x & 31;
    const int rl = threadIdx.x >> 5;
    const int stride = gridDim.x * 8;
    float4 s = {0.f,0.f,0.f,0.f}, q = {0.f,0.f,0.f,0.f};
    for (int e = blockIdx.x * 8 + rl; e < EE; e += stride) {
        if (e + stride < EE)
            __builtin_prefetch(&W1e[(size_t)(e + stride) * 128 + cg * 4], 0, 1);
        int si = src[e], di = dst[e];
        float4 a = reinterpret_cast<const float4*>(W1e)[e * 32 + cg];
        float4 b = reinterpret_cast<const float4*>(W2h)[si * 32 + cg];
        float4 d = reinterpret_cast<const float4*>(W3h)[di * 32 + cg];
        float x0 = a.x + b.x + d.x, x1 = a.y + b.y + d.y;
        float x2 = a.z + b.z + d.z, x3 = a.w + b.w + d.w;
        s.x += x0; s.y += x1; s.z += x2; s.w += x3;
        q.x += x0 * x0; q.y += x1 * x1; q.z += x2 * x2; q.w += x3 * x3;
    }
    atomicAdd(&stats[cg * 4 + 0], s.x); atomicAdd(&stats[cg * 4 + 1], s.y);
    atomicAdd(&stats[cg * 4 + 2], s.z); atomicAdd(&stats[cg * 4 + 3], s.w);
    atomicAdd(&stats[128 + cg * 4 + 0], q.x); atomicAdd(&stats[128 + cg * 4 + 1], q.y);
    atomicAdd(&stats[128 + cg * 4 + 2], q.z); atomicAdd(&stats[128 + cg * 4 + 3], q.w);
}

__global__ __launch_bounds__(256) void edge_apply(
    const float* __restrict__ e0, const float* __restrict__ W1e,
    const float* __restrict__ W2h, const float* __restrict__ W3h,
    const int* __restrict__ src, const int* __restrict__ dst,
    const float* __restrict__ stats,
    const float* __restrict__ gamma_e, const float* __restrict__ beta_e,
    float* __restrict__ e_out, unsigned short* __restrict__ sig)
{
    const float invE = 1.f / (float)EE;
    const int total = EE * 32;
    for (int idx = blockIdx.x * blockDim.x + threadIdx.x; idx < total;
         idx += gridDim.x * blockDim.x) {
        int e = idx >> 5, cg = idx & 31;
        int si = src[e], di = dst[e];
        float4 a = reinterpret_cast<const float4*>(W1e)[idx];
        float4 b = reinterpret_cast<const float4*>(W2h)[si * 32 + cg];
        float4 d = reinterpret_cast<const float4*>(W3h)[di * 32 + cg];
        float4 z = reinterpret_cast<const float4*>(e0)[idx];
        float o[4]; unsigned short sb[4];
#pragma unroll
        for (int j = 0; j < 4; ++j) {
            int col  = cg * 4 + j;
            float x  = getc(a, j) + getc(b, j) + getc(d, j);
            float mu = stats[col] * invE;
            float var = stats[128 + col] * invE - mu * mu;
            float bn = (x - mu) * rsqrtf(var + 1e-5f) * gamma_e[col] + beta_e[col];
            float en = getc(z, j) + fmaxf(bn, 0.f);
            o[j]  = en;
            sb[j] = f2bf(1.f / (1.f + __expf(-en)));
        }
        float4 ov = {o[0], o[1], o[2], o[3]};
        reinterpret_cast<float4*>(e_out)[idx] = ov;
        uint2 p;
        p.x = (unsigned int)sb[0] | ((unsigned int)sb[1] << 16);
        p.y = (unsigned int)sb[2] | ((unsigned int)sb[3] << 16);
        reinterpret_cast<uint2*>(sig)[idx] = p;
    }
}

__global__ __launch_bounds__(256) void node_stats(
    const float* __restrict__ Uh, const float* __restrict__ num,
    const float* __restrict__ den, float* __restrict__ stats)
{
    const int total = NN * 32;
    for (int idx = blockIdx.x * blockDim.x + threadIdx.x; idx < total;
         idx += gridDim.x * blockDim.x) {
        int cg = idx & 31;
        float4 u  = reinterpret_cast<const float4*>(Uh)[idx];
        float4 nm = reinterpret_cast<const float4*>(num)[idx];
        float4 dn = reinterpret_cast<const float4*>(den)[idx];
#pragma unroll
        for (int j = 0; j < 4; ++j) {
            float x = getc(u, j) + getc(nm, j) / (getc(dn, j) + 1e-5f);
            atomicAdd(&stats[256 + cg * 4 + j], x);
            atomicAdd(&stats[384 + cg * 4 + j], x * x);
        }
    }
}

__global__ __launch_bounds__(256) void node_apply(
    const float* __restrict__ h0, const float* __restrict__ Uh,
    const float* __restrict__ num, const float* __restrict__ den,
    const float* __restrict__ stats,
    const float* __restrict__ gamma_h, const float* __restrict__ beta_h,
    float* __restrict__ out)
{
    const float invN = 1.f / (float)NN;
    const int total = NN * 32;
    for (int idx = blockIdx.x * blockDim.x + threadIdx.x; idx < total;
         idx += gridDim.x * blockDim.x) {
        int cg = idx & 31;
        float4 z  = reinterpret_cast<const float4*>(h0)[idx];
        float4 u  = reinterpret_cast<const float4*>(Uh)[idx];
        float4 nm = reinterpret_cast<const float4*>(num)[idx];
        float4 dn = reinterpret_cast<const float4*>(den)[idx];
        float o[4];
#pragma unroll
        for (int j = 0; j < 4; ++j) {
            int col  = cg * 4 + j;
            float x  = getc(u, j) + getc(nm, j) / (getc(dn, j) + 1e-5f);
            float mu = stats[256 + col] * invN;
            float var = stats[384 + col] * invN - mu * mu;
            float bn = (x - mu) * rsqrtf(var + 1e-5f) * gamma_h[col] + beta_h[col];
            o[j] = getc(z, j) + fmaxf(bn, 0.f);
        }
        float4 ov = {o[0], o[1], o[2], o[3]};
        reinterpret_cast<float4*>(out)[idx] = ov;
    }
}

// ---------------------------------------------------------------------------
extern "C" void kernel_launch(void* const* d_in, const int* in_sizes, int n_in,
                              void* d_out, int out_size, void* d_ws, size_t ws_size,
                              hipStream_t stream)
{
    const float* h      = (const float*)d_in[0];
    const float* e      = (const float*)d_in[1];
    const int*   src    = (const int*)d_in[2];
    const int*   dst    = (const int*)d_in[3];
    const float* W_node = (const float*)d_in[4];
    const float* W_edge = (const float*)d_in[5];
    const float* W_eta  = (const float*)d_in[6];
    const float* Uw = (const float*)d_in[7];  const float* Ub = (const float*)d_in[8];
    const float* Vw = (const float*)d_in[9];  const float* Vb = (const float*)d_in[10];
    const float* W1w = (const float*)d_in[11]; const float* W1b = (const float*)d_in[12];
    const float* W2w = (const float*)d_in[13]; const float* W2b = (const float*)d_in[14];
    const float* W3w = (const float*)d_in[15]; const float* W3b = (const float*)d_in[16];
    const float* gamma_h = (const float*)d_in[17]; const float* beta_h = (const float*)d_in[18];
    const float* gamma_e = (const float*)d_in[19]; const float* beta_e = (const float*)d_in[20];

    char* ws = (char*)d_ws;
    size_t off = 0;
    auto alloc = [&](size_t bytes) -> void* {
        off = (off + 255) & ~(size_t)255;
        void* p = ws + off;
        off += bytes;
        return p;
    };
    const size_t nd  = (size_t)NN * DDIM;     // logical node rows
    const size_t ndp = (size_t)NPAD * DDIM;   // padded (unguarded GEMM stores)
    const size_t ed  = (size_t)EE * DDIM;
    float* h0    = (float*)alloc(ndp * 4);
    float* Uh    = (float*)alloc(ndp * 4);
    float* Vh    = (float*)alloc(ndp * 4);
    float* W2h   = (float*)alloc(ndp * 4);
    float* W3h   = (float*)alloc(ndp * 4);
    float* numB  = (float*)alloc(nd * 4);
    float* denB  = (float*)alloc(nd * 4);
    float* stats = (float*)alloc(640 * 4);    // 512 stats + 128 zero-bias
    float* zbias = stats + 512;
    unsigned short* wsw = (unsigned short*)alloc((size_t)8 * 16384 * 2); // packed weights
    float* e0    = (float*)alloc(ed * 4);
    float* W1e   = (float*)alloc(ed * 4);
    unsigned short* sig = (unsigned short*)alloc(ed * 2);
    (void)ws_size; (void)in_sizes; (void)n_in; (void)out_size;

    float* h_out = (float*)d_out;            // [N*D]
    float* e_out = (float*)d_out + nd;       // [E*D]

    dim3 blk(256);

    zero_buf<<<2, 512, 0, stream>>>(stats, 640);

    // pack all 8 weight matrices: bf16, transposed, XOR-swizzled
    Ptr8 wp = {{ W_node, Uw, Vw, W2w, W3w, W_edge, W1w, W_eta }};
    pack_weights<<<8, blk, 0, stream>>>(wp, wsw);
    const unsigned short* B_node = wsw + 0 * 16384;
    const unsigned short* B_U    = wsw + 1 * 16384;
    const unsigned short* B_V    = wsw + 2 * 16384;
    const unsigned short* B_W2   = wsw + 3 * 16384;
    const unsigned short* B_W3   = wsw + 4 * 16384;
    const unsigned short* B_edge = wsw + 5 * 16384;
    const unsigned short* B_W1   = wsw + 6 * 16384;
    const unsigned short* B_eta  = wsw + 7 * 16384;

    // node-side linears: one multi-job GEMM (A = h read once per block)
    GemmJobs nodeJobs = {};
    nodeJobs.j[0] = {B_node, zbias, h0};
    nodeJobs.j[1] = {B_U,  Ub,  Uh};
    nodeJobs.j[2] = {B_V,  Vb,  Vh};
    nodeJobs.j[3] = {B_W2, W2b, W2h};
    nodeJobs.j[4] = {B_W3, W3b, W3h};
    gemm128_multi<5><<<dim3(NPAD / 128), blk, 0, stream>>>(h, nodeJobs, NN);

    // edge-side linears: e read from HBM exactly once
    GemmJobs edgeJobs = {};
    edgeJobs.j[0] = {B_edge, zbias, e0};
    edgeJobs.j[1] = {B_W1, W1b, W1e};
    gemm128_multi<2><<<dim3(EE / 128), blk, 0, stream>>>(e, edgeJobs, EE);

    // BN stats over pre, then e_new + sigmoid(e_new)->bf16
    edge_stats<<<2048, blk, 0, stream>>>(W1e, W2h, W3h, src, dst, stats);
    edge_apply<<<4096, blk, 0, stream>>>(e0, W1e, W2h, W3h, src, dst, stats,
                                         gamma_e, beta_e, e_out, sig);

    // sigma GEMM fused with gated segment reduction (fully async tiles)
    sigma_gemm_reduce<<<dim3(EE / 128), blk, 0, stream>>>(sig, B_eta, Vh, src,
                                                          numB, denB);

    // node BN stats + final h_new
    node_stats<<<512, blk, 0, stream>>>(Uh, numB, denB, stats);
    node_apply<<<512, blk, 0, stream>>>(h0, Uh, numB, denB, stats,
                                        gamma_h, beta_h, h_out);
}
